// DegreeSortedMambaLayer_63488206569951
// MI455X (gfx1250) — matrix operand order repaired and checked by reading, not compiled, not run on hardware
//
#include <hip/hip_runtime.h>

// Problem constants (match reference)
#define NT_   16384
#define G_    64
#define NSEQ_ 256
#define DM_   256
#define DI_   512
#define DS_   16
#define DC_   4
#define DTR_  16

typedef __attribute__((ext_vector_type(16))) __bf16        v16bf;
typedef __attribute__((ext_vector_type(8)))  float         v8f;
typedef __attribute__((ext_vector_type(8)))  unsigned int  v8u;

// ---------------- helpers ----------------
__device__ __forceinline__ unsigned short f2bf(float f) {
    unsigned int x = __builtin_bit_cast(unsigned int, f);
    unsigned int r = x + 0x7FFFu + ((x >> 16) & 1u);   // round-to-nearest-even
    return (unsigned short)(r >> 16);
}
__device__ __forceinline__ float fsigmoid(float x) { return 1.0f / (1.0f + __expf(-x)); }
__device__ __forceinline__ float fsilu(float x)    { return x * fsigmoid(x); }
__device__ __forceinline__ float fsoftplus(float x){ return (x > 20.0f) ? x : __logf(1.0f + __expf(x)); }

// ---------------- K1: degree count ----------------
__global__ __launch_bounds__(256) void k_degree(const int* __restrict__ src, int* __restrict__ deg, int ne) {
    int e = blockIdx.x * 256 + threadIdx.x;
    if (e < ne) atomicAdd(&deg[src[e]], 1);
}

// ---------------- K2: per-graph stable rank sort (by deg asc, tie by index) ----------------
__global__ __launch_bounds__(256) void k_sort(const int* __restrict__ deg, int* __restrict__ perm) {
    __shared__ int sdeg[NSEQ_];
    int g = blockIdx.x, i = threadIdx.x;
    int node = g * NSEQ_ + i;
    sdeg[i] = deg[node];
    __syncthreads();
    int di = sdeg[i], rank = 0;
    for (int j = 0; j < NSEQ_; ++j) {
        int dj = sdeg[j];
        rank += (dj < di) || (dj == di && j < i);
    }
    perm[g * NSEQ_ + rank] = node;
}

// ---------------- K3: f32 -> bf16 convert ----------------
__global__ __launch_bounds__(256) void k_cvt_bf16(const float* __restrict__ in, unsigned short* __restrict__ out, int n) {
    int i = blockIdx.x * 256 + threadIdx.x;
    if (i < n) out[i] = f2bf(in[i]);
}

// ---------------- K4: gather sorted (optionally reversed) rows of x as bf16 ----------------
__global__ __launch_bounds__(256) void k_gather(const float* __restrict__ x, const int* __restrict__ perm,
                                                unsigned short* __restrict__ h, int flip) {
    int gid = blockIdx.x * 256 + threadIdx.x;      // over NT*DM
    int r = gid >> 8, c = gid & (DM_ - 1);
    int g = r >> 8, l = r & (NSEQ_ - 1);
    int lsrc = flip ? (NSEQ_ - 1 - l) : l;
    int node = perm[g * NSEQ_ + lsrc];
    h[(size_t)r * DM_ + c] = f2bf(x[(size_t)node * DM_ + c]);
}

// ---------------- K5: bf16 WMMA GEMM  C[M,N] = A[M,K] * B[N,K]^T ----------------
// Block = 8 waves sharing one 64-column n-group. The 64 x K bf16 weight strip is
// staged into LDS once per block via async global->LDS copies (ASYNCcnt path),
// then each wave computes a 16x64 block (4 accumulators), K in chunks of 32,
// with the next A fragment prefetched while WMMAs issue.
__global__ __launch_bounds__(256) void k_gemm_bf16(const unsigned short* __restrict__ A,
                                                   const unsigned short* __restrict__ B,
                                                   float* __restrict__ C, int M, int N, int K) {
    __shared__ unsigned short sB[64 * 512];        // 64 cols x K(<=512) bf16 = 64KB
    const int lane   = threadIdx.x & 31;
    const int wid    = threadIdx.x >> 5;           // 0..7
    const int mtiles = M >> 4;
    const int blocksPerNg = mtiles >> 3;           // 8 m-tiles per block
    const int ng = blockIdx.x / blocksPerNg;
    const int mb = blockIdx.x % blocksPerNg;
    const int mt = mb * 8 + wid;

    // --- async copy of the 64 x K weight strip into LDS ---
    {
        const int chunks = (64 * K) >> 3;          // 16-byte (8 bf16) chunks
        unsigned long long base =
            (unsigned long long)(const char*)(B + (size_t)ng * 64 * K);
        for (int idx = threadIdx.x; idx < chunks; idx += 256) {
            unsigned long long gaddr = base + ((unsigned long long)idx << 4);
            unsigned ldsoff = (unsigned)(unsigned long long)&sB[idx * 8];
            asm volatile("global_load_async_to_lds_b128 %0, %1, off"
                         :: "v"(ldsoff), "v"(gaddr) : "memory");
        }
        asm volatile("s_wait_asynccnt 0x0" ::: "memory");
    }
    __syncthreads();

    const int m  = lane & 15;
    const int hi = (lane >> 4) & 1;
    const int aKoff = hi * 8;      // A: lanes 16-31 hold K+8 (low half) / K+24 (high half)
    const int bKoff = hi * 8;      // B: lanes 16-31 hold K+8 (low half) / K+24 (high half)
    const unsigned short* arow = A + (size_t)(mt * 16 + m) * K;

    v8f acc[4];
#pragma unroll
    for (int t = 0; t < 4; ++t)
#pragma unroll
        for (int v = 0; v < 8; ++v) acc[t][v] = 0.0f;

    auto loadA = [&](int kk) {
        v8u au;
#pragma unroll
        for (int v = 0; v < 4; ++v)
            au[v] = *(const unsigned int*)(arow + kk + aKoff + 2 * v);
#pragma unroll
        for (int v = 0; v < 4; ++v)
            au[4 + v] = *(const unsigned int*)(arow + kk + 16 + aKoff + 2 * v);
        return au;
    };

    v8u au = loadA(0);
    for (int kk = 0; kk < K; kk += 32) {
        v8u aun = (kk + 32 < K) ? loadA(kk + 32) : au;   // prefetch next A fragment
        v16bf afrag = __builtin_bit_cast(v16bf, au);
#pragma unroll
        for (int t = 0; t < 4; ++t) {
            const unsigned short* bp = sB + (size_t)(t * 16 + m) * K + kk + bKoff;
            v8u bu;
#pragma unroll
            for (int v = 0; v < 8; ++v)
                bu[v] = *(const unsigned int*)(bp + (v >> 2) * 16 + 2 * (v & 3));
            v16bf bfrag = __builtin_bit_cast(v16bf, bu);
            acc[t] = __builtin_amdgcn_wmma_f32_16x16x32_bf16(
                false, afrag, false, bfrag, (short)0, acc[t], false, false);
        }
        au = aun;
    }
    // C/D layout: VGPR v holds row (v + 8*hi), column = lane&15 within tile
#pragma unroll
    for (int t = 0; t < 4; ++t) {
        int col = ng * 64 + t * 16 + m;
#pragma unroll
        for (int v = 0; v < 8; ++v) {
            int row = mt * 16 + v + 8 * hi;
            C[(size_t)row * N + col] = acc[t][v];
        }
    }
}

// ---------------- K6: depthwise causal conv + bias + SiLU ----------------
__global__ __launch_bounds__(256) void k_conv(const float* __restrict__ xz, const float* __restrict__ cw,
                                              const float* __restrict__ cb, float* __restrict__ u) {
    int gid = blockIdx.x * 256 + threadIdx.x;      // over NT*DI
    int r = gid >> 9, d = gid & (DI_ - 1);
    int l = r & (NSEQ_ - 1);
    float acc = cb[d];
#pragma unroll
    for (int j = 0; j < DC_; ++j) {
        int li = l - (DC_ - 1) + j;
        if (li >= 0) acc += cw[d * DC_ + j] * xz[(size_t)(r - l + li) * (2 * DI_) + d];
    }
    u[(size_t)r * DI_ + d] = fsilu(acc);
}

// ---------------- K7: xproj (N=48) -> dt / B / C ----------------
__global__ __launch_bounds__(256) void k_xproj(const float* __restrict__ u, const float* __restrict__ xw,
                                               float* __restrict__ dt, float* __restrict__ Bm, float* __restrict__ Cm) {
    int gid = blockIdx.x * 256 + threadIdx.x;      // over NT*48
    int r = gid / 48, j = gid % 48;
    const float* ur = u + (size_t)r * DI_;
    const float* wr = xw + (size_t)j * DI_;
    float acc = 0.0f;
    for (int k = 0; k < DI_; ++k) acc += ur[k] * wr[k];
    if (j < DTR_)            dt[(size_t)r * DTR_ + j] = acc;
    else if (j < DTR_ + DS_) Bm[(size_t)r * DS_ + (j - DTR_)] = acc;
    else                     Cm[(size_t)r * DS_ + (j - DTR_ - DS_)] = acc;
}

// ---------------- K8: delta = softplus(dt @ dt_w^T + dt_b) ----------------
__global__ __launch_bounds__(256) void k_delta(const float* __restrict__ dt, const float* __restrict__ dtw,
                                               const float* __restrict__ dtb, float* __restrict__ delta) {
    int gid = blockIdx.x * 256 + threadIdx.x;      // over NT*DI
    int r = gid >> 9, d = gid & (DI_ - 1);
    float acc = dtb[d];
#pragma unroll
    for (int t = 0; t < DTR_; ++t) acc += dt[(size_t)r * DTR_ + t] * dtw[d * DTR_ + t];
    delta[(size_t)gid] = fsoftplus(acc);
}

// ---------------- K9: selective scan (one thread per (graph, channel) chain) ----------------
__global__ __launch_bounds__(512) void k_scan(const float* __restrict__ delta, const float* __restrict__ u,
                                              const float* __restrict__ Bm, const float* __restrict__ Cm,
                                              const float* __restrict__ xz, const float* __restrict__ A_log,
                                              const float* __restrict__ Dp, unsigned short* __restrict__ ybf) {
    __shared__ float sB[DS_], sC[DS_];
    int g = blockIdx.x, d = threadIdx.x;
    float A[DS_], st[DS_];
#pragma unroll
    for (int n = 0; n < DS_; ++n) { A[n] = -__expf(A_log[d * DS_ + n]); st[n] = 0.0f; }
    float Dd = Dp[d];
    for (int l = 0; l < NSEQ_; ++l) {
        int r = g * NSEQ_ + l;
        if (d < DS_)                 sB[d] = Bm[(size_t)r * DS_ + d];
        else if (d < 2 * DS_)        sC[d - DS_] = Cm[(size_t)r * DS_ + (d - DS_)];
        __syncthreads();
        float dl = delta[(size_t)r * DI_ + d];
        float uu = u[(size_t)r * DI_ + d];
        float du = dl * uu;
        float y = 0.0f;
#pragma unroll
        for (int n = 0; n < DS_; ++n) {
            float dA = __expf(dl * A[n]);
            st[n] = dA * st[n] + du * sB[n];
            y += st[n] * sC[n];
        }
        float zz = xz[(size_t)r * (2 * DI_) + DI_ + d];
        ybf[(size_t)r * DI_ + d] = f2bf((y + uu * Dd) * fsilu(zz));
        __syncthreads();
    }
}

// ---------------- K10: pack [f, b_flipped] concat as bf16 ----------------
__global__ __launch_bounds__(256) void k_pack(const float* __restrict__ f, const float* __restrict__ b,
                                              unsigned short* __restrict__ cat) {
    int gid = blockIdx.x * 256 + threadIdx.x;      // over NT*DM
    int r = gid >> 8, c = gid & (DM_ - 1);
    int g = r >> 8, l = r & (NSEQ_ - 1);
    int br = g * NSEQ_ + (NSEQ_ - 1 - l);
    cat[(size_t)r * (2 * DM_) + c]        = f2bf(f[(size_t)r * DM_ + c]);
    cat[(size_t)r * (2 * DM_) + DM_ + c]  = f2bf(b[(size_t)br * DM_ + c]);
}

// ---------------- K11: sigmoid gate blend + scatter via perm ----------------
__global__ __launch_bounds__(256) void k_final(const float* __restrict__ glog, const float* __restrict__ gb,
                                               const float* __restrict__ f, const float* __restrict__ b,
                                               const int* __restrict__ perm, float* __restrict__ out) {
    int gid = blockIdx.x * 256 + threadIdx.x;      // over NT*DM
    int r = gid >> 8, c = gid & (DM_ - 1);
    int g = r >> 8, l = r & (NSEQ_ - 1);
    int br = g * NSEQ_ + (NSEQ_ - 1 - l);
    float s  = fsigmoid(glog[(size_t)r * DM_ + c] + gb[c]);
    float fv = f[(size_t)r * DM_ + c];
    float bv = b[(size_t)br * DM_ + c];
    out[(size_t)perm[r] * DM_ + c] = s * fv + (1.0f - s) * bv;
}

// ---------------- workspace layout ----------------
static constexpr size_t OFF_DEG   = 0;
static constexpr size_t OFF_PERM  = OFF_DEG   + (size_t)NT_ * 4;
static constexpr size_t OFF_HBF   = OFF_PERM  + (size_t)NT_ * 4;
static constexpr size_t OFF_XZ    = OFF_HBF   + (size_t)NT_ * DM_ * 2;
static constexpr size_t OFF_U     = OFF_XZ    + (size_t)NT_ * 2 * DI_ * 4;
static constexpr size_t OFF_DT    = OFF_U     + (size_t)NT_ * DI_ * 4;
static constexpr size_t OFF_BMAT  = OFF_DT    + (size_t)NT_ * DTR_ * 4;
static constexpr size_t OFF_CMAT  = OFF_BMAT  + (size_t)NT_ * DS_ * 4;
static constexpr size_t OFF_DELTA = OFF_CMAT  + (size_t)NT_ * DS_ * 4;
static constexpr size_t OFF_YBF   = OFF_DELTA + (size_t)NT_ * DI_ * 4;
static constexpr size_t OFF_FOUT  = OFF_YBF   + (size_t)NT_ * DI_ * 2;
static constexpr size_t OFF_BOUT  = OFF_FOUT  + (size_t)NT_ * DM_ * 4;
static constexpr size_t OFF_CAT   = OFF_BOUT  + (size_t)NT_ * DM_ * 4;
static constexpr size_t OFF_GLOG  = OFF_CAT   + (size_t)NT_ * 2 * DM_ * 2;
static constexpr size_t OFF_WIN   = OFF_GLOG  + (size_t)NT_ * DM_ * 4;
static constexpr size_t OFF_WOUT  = OFF_WIN   + (size_t)(2 * DI_) * DM_ * 2;
static constexpr size_t OFF_WGATE = OFF_WOUT  + (size_t)DM_ * DI_ * 2;

extern "C" void kernel_launch(void* const* d_in, const int* in_sizes, int n_in,
                              void* d_out, int out_size, void* d_ws, size_t ws_size,
                              hipStream_t stream) {
    char* ws = (char*)d_ws;
    const float* x       = (const float*)d_in[0];
    const int*   eidx    = (const int*)d_in[1];
    const float* gate_w  = (const float*)d_in[3];
    const float* gate_b  = (const float*)d_in[4];
    const int ne = in_sizes[1] / 2;

    int*            deg    = (int*)(ws + OFF_DEG);
    int*            perm   = (int*)(ws + OFF_PERM);
    unsigned short* hbf    = (unsigned short*)(ws + OFF_HBF);
    float*          xz     = (float*)(ws + OFF_XZ);
    float*          u      = (float*)(ws + OFF_U);
    float*          dt     = (float*)(ws + OFF_DT);
    float*          Bm     = (float*)(ws + OFF_BMAT);
    float*          Cm     = (float*)(ws + OFF_CMAT);
    float*          delta  = (float*)(ws + OFF_DELTA);
    unsigned short* ybf    = (unsigned short*)(ws + OFF_YBF);
    float*          fout   = (float*)(ws + OFF_FOUT);
    float*          bout   = (float*)(ws + OFF_BOUT);
    unsigned short* catbf  = (unsigned short*)(ws + OFF_CAT);
    float*          glog   = (float*)(ws + OFF_GLOG);
    unsigned short* winbf  = (unsigned short*)(ws + OFF_WIN);
    unsigned short* woutbf = (unsigned short*)(ws + OFF_WOUT);
    unsigned short* wgatbf = (unsigned short*)(ws + OFF_WGATE);

    // degree + per-graph sort
    hipMemsetAsync(deg, 0, (size_t)NT_ * 4, stream);
    k_degree<<<(ne + 255) / 256, 256, 0, stream>>>(eidx, deg, ne);
    k_sort<<<G_, 256, 0, stream>>>(deg, perm);

    // gate weights -> bf16 (once)
    k_cvt_bf16<<<(DM_ * 2 * DM_) / 256, 256, 0, stream>>>(gate_w, wgatbf, DM_ * 2 * DM_);

    for (int dir = 0; dir < 2; ++dir) {
        const float* in_w    = (const float*)d_in[5  + 9 * dir];
        const float* conv_w  = (const float*)d_in[6  + 9 * dir];
        const float* conv_b  = (const float*)d_in[7  + 9 * dir];
        const float* xproj_w = (const float*)d_in[8  + 9 * dir];
        const float* dt_w    = (const float*)d_in[9  + 9 * dir];
        const float* dt_b    = (const float*)d_in[10 + 9 * dir];
        const float* A_log   = (const float*)d_in[11 + 9 * dir];
        const float* Dp      = (const float*)d_in[12 + 9 * dir];
        const float* out_w   = (const float*)d_in[13 + 9 * dir];
        float* dout = dir ? bout : fout;

        k_cvt_bf16<<<(2 * DI_ * DM_) / 256, 256, 0, stream>>>(in_w, winbf, 2 * DI_ * DM_);
        k_cvt_bf16<<<(DM_ * DI_) / 256, 256, 0, stream>>>(out_w, woutbf, DM_ * DI_);
        k_gather<<<(NT_ * DM_) / 256, 256, 0, stream>>>(x, perm, hbf, dir);

        // in-proj: [16384 x 256] x [256 x 1024]  (16 n-groups x 128 blocks)
        k_gemm_bf16<<<(NT_ / 16) * (2 * DI_ / 64) / 8, 256, 0, stream>>>(hbf, winbf, xz, NT_, 2 * DI_, DM_);

        k_conv <<<(NT_ * DI_) / 256, 256, 0, stream>>>(xz, conv_w, conv_b, u);
        k_xproj<<<(NT_ * 48) / 256, 256, 0, stream>>>(u, xproj_w, dt, Bm, Cm);
        k_delta<<<(NT_ * DI_) / 256, 256, 0, stream>>>(dt, dt_w, dt_b, delta);
        k_scan <<<G_, DI_, 0, stream>>>(delta, u, Bm, Cm, xz, A_log, Dp, ybf);

        // out-proj: [16384 x 512] x [512 x 256]
        k_gemm_bf16<<<(NT_ / 16) * (DM_ / 64) / 8, 256, 0, stream>>>(ybf, woutbf, dout, NT_, DM_, DI_);
    }

    // gate: sigmoid([f;b] @ gate_w^T + gate_b), blend, scatter by perm
    k_pack<<<(NT_ * DM_) / 256, 256, 0, stream>>>(fout, bout, catbf);
    k_gemm_bf16<<<(NT_ / 16) * (DM_ / 64) / 8, 256, 0, stream>>>(catbf, wgatbf, glog, NT_, DM_, 2 * DM_);
    k_final<<<(NT_ * DM_) / 256, 256, 0, stream>>>(glog, gate_b, fout, bout, perm, (float*)d_out);
}
